// DualAttentionModule_67379446939965
// MI455X (gfx1250) — compile-verified, matrix operand order checked by reading, not run on hardware
//
#include <hip/hip_runtime.h>

// ---------------------------------------------------------------------------
// CDNA5 (gfx1250) dual-attention transformer block, wave32 + v_wmma_f32_16x16x32_f16
// ---------------------------------------------------------------------------

typedef __attribute__((ext_vector_type(16))) _Float16 v16h;
typedef __attribute__((ext_vector_type(8)))  float    v8f;

#define WMMA16(A, B, C) \
  __builtin_amdgcn_wmma_f32_16x16x32_f16(false, (A), false, (B), (short)0, (C), false, false)

// ---- WMMA operand loaders (layouts per CDNA5 ISA §7.12.2) ----
// A (16x32, f16): lane m=l&15 holds K = kh..kh+7 and kh+16..kh+23, kh = (l>>4)*8
__device__ __forceinline__ v16h load_a_f32(const float* __restrict__ p0, int ld,
                                           int kbase, int lane) {
  const int r  = lane & 15;
  const int kh = (lane >> 4) << 3;
  const float* p = p0 + (size_t)r * ld + kbase + kh;
  v16h a;
#pragma unroll
  for (int j = 0; j < 8; ++j) {
    a[j]     = (_Float16)p[j];
    a[8 + j] = (_Float16)p[16 + j];
  }
  return a;
}

// B from f16 transposed weights Wt(N,K): lane holds column n=l&15 -> row of Wt
__device__ __forceinline__ v16h load_b_h(const _Float16* __restrict__ p0, int ldk,
                                         int kbase, int lane) {
  const int n  = lane & 15;
  const int kh = (lane >> 4) << 3;
  const _Float16* p = p0 + (size_t)n * ldk + kbase + kh;
  v16h b;
#pragma unroll
  for (int j = 0; j < 8; ++j) {
    b[j]     = p[j];
    b[8 + j] = p[16 + j];
  }
  return b;
}

// A from f16 LDS tile (same pattern as load_a_f32)
__device__ __forceinline__ v16h load_a_h(const _Float16* __restrict__ p0, int ld,
                                         int kbase, int lane) {
  const int r  = lane & 15;
  const int kh = (lane >> 4) << 3;
  const _Float16* p = p0 + (size_t)r * ld + kbase + kh;
  v16h a;
#pragma unroll
  for (int j = 0; j < 8; ++j) {
    a[j]     = p[j];
    a[8 + j] = p[16 + j];
  }
  return a;
}

// B from f32 row-major (K,N) matrix: column per lane (strided)
__device__ __forceinline__ v16h load_b_col_f32(const float* __restrict__ p0, int ld,
                                               int kbase, int lane) {
  const int n  = lane & 15;
  const int kh = (lane >> 4) << 3;
  const float* p = p0 + (size_t)(kbase + kh) * ld + n;
  v16h b;
#pragma unroll
  for (int j = 0; j < 8; ++j) {
    b[j]     = (_Float16)p[(size_t)j * ld];
    b[8 + j] = (_Float16)p[(size_t)(16 + j) * ld];
  }
  return b;
}

// ---------------------------------------------------------------------------
// Weight convert: Wt[n*K+k] = (f16)W[k*N+n]
__global__ void transpose_w_kernel(const float* __restrict__ W, _Float16* __restrict__ Wt,
                                   int K, int N) {
  int idx = blockIdx.x * 256 + threadIdx.x;
  if (idx >= K * N) return;
  int n = idx / K, k = idx - n * K;
  Wt[idx] = (_Float16)W[(size_t)k * N + n];
}

// x (B,F,T,D) -> xf chunk (T*F, D) for batch b
__global__ void gather_feat_kernel(const float* __restrict__ x, float* __restrict__ xf, int b) {
  int row = blockIdx.x;          // row = t*64 + f
  int t = row >> 6, f = row & 63;
  int c = threadIdx.x;
  xf[(size_t)row * 256 + c] = x[(((size_t)b * 64 + f) * 128 + t) * 256 + c];
}

// feature output (T*F, D) -> xt chunk (F*T, D)
__global__ void scatter_feat_kernel(const float* __restrict__ x2, float* __restrict__ xt) {
  int row = blockIdx.x;          // row = t*64 + f
  int t = row >> 6, f = row & 63;
  int c = threadIdx.x;
  xt[((size_t)f * 128 + t) * 256 + c] = x2[(size_t)row * 256 + c];
}

// ---------------------------------------------------------------------------
// Generic GEMM + bias: C(M,N) = X(M,K) @ W(K,N) + bias. One wave -> 16x64 tile.
__global__ void __launch_bounds__(32)
gemm_bias_kernel(const float* __restrict__ X, const _Float16* __restrict__ Wt,
                 const float* __restrict__ bias, float* __restrict__ C, int K, int N) {
  const int lane = threadIdx.x;
  const int mt = blockIdx.x, n0 = blockIdx.y * 64;
  const float* Xp = X + (size_t)mt * 16 * K;
  v8f acc[4] = {{0,0,0,0,0,0,0,0},{0,0,0,0,0,0,0,0},{0,0,0,0,0,0,0,0},{0,0,0,0,0,0,0,0}};
  for (int k = 0; k < K; k += 32) {
    v16h a = load_a_f32(Xp, K, k, lane);
#pragma unroll
    for (int t = 0; t < 4; ++t) {
      v16h b = load_b_h(Wt + (size_t)(n0 + t * 16) * K, K, k, lane);
      acc[t] = WMMA16(a, b, acc[t]);
    }
  }
  const int rb = (lane >> 4) << 3, col = lane & 15;
#pragma unroll
  for (int t = 0; t < 4; ++t) {
    int cc = n0 + t * 16 + col;
    float bv = bias[cc];
#pragma unroll
    for (int r = 0; r < 8; ++r)
      C[((size_t)mt * 16 + rb + r) * N + cc] = acc[t][r] + bv;
  }
}

// Scores: P = Q @ K^T * 1/sqrt(32), written to the softmax-weights output region.
// grid (nl*8+h, S/16), 1 wave. GQA: kv head = h>>1.
__global__ void __launch_bounds__(32)
attn_scores_kernel(const float* __restrict__ Q, const float* __restrict__ Kb,
                   float* __restrict__ W, int S) {
  const int lane = threadIdx.x;
  const int nl = blockIdx.x >> 3, h = blockIdx.x & 7, mt = blockIdx.y;
  const float* Qp = Q + ((size_t)nl * S + mt * 16) * 256 + h * 32;
  const float* Kp = Kb + ((size_t)nl * S) * 128 + (h >> 1) * 32;
  float* Wp = W + (((size_t)(nl * 8 + h)) * S + mt * 16) * S;
  v16h a = load_a_f32(Qp, 256, 0, lane);
  const int rb = (lane >> 4) << 3, col = lane & 15;
  for (int nt = 0; nt < (S >> 4); ++nt) {
    v16h b = load_a_f32(Kp + (size_t)nt * 16 * 128, 128, 0, lane);  // B = K^T
    v8f acc = {0,0,0,0,0,0,0,0};
    acc = WMMA16(a, b, acc);
#pragma unroll
    for (int r = 0; r < 8; ++r)
      Wp[(size_t)(rb + r) * S + nt * 16 + col] = acc[r] * 0.17677669529663687f;
  }
}

// In-place row softmax, one wave per row (S = 64 or 128).
__global__ void __launch_bounds__(256)
softmax_kernel(float* __restrict__ W, int S) {
  const int w = threadIdx.x >> 5, lane = threadIdx.x & 31;
  float* p = W + ((size_t)blockIdx.x * 8 + w) * S;
  const int nv = S >> 5;
  float vals[4];
  float mx = -3.0e38f;
  for (int i = 0; i < nv; ++i) { vals[i] = p[lane + (i << 5)]; mx = fmaxf(mx, vals[i]); }
  for (int off = 16; off > 0; off >>= 1) mx = fmaxf(mx, __shfl_xor(mx, off));
  float sum = 0.f;
  for (int i = 0; i < nv; ++i) { vals[i] = __expf(vals[i] - mx); sum += vals[i]; }
  for (int off = 16; off > 0; off >>= 1) sum += __shfl_xor(sum, off);
  float inv = 1.f / sum;
  for (int i = 0; i < nv; ++i) p[lane + (i << 5)] = vals[i] * inv;
}

// O = P @ V, head-interleaved output (token, h*32+j). grid (nl*8+h, S/16), 1 wave.
__global__ void __launch_bounds__(32)
attn_out_kernel(const float* __restrict__ W, const float* __restrict__ V,
                float* __restrict__ O, int S) {
  const int lane = threadIdx.x;
  const int nl = blockIdx.x >> 3, h = blockIdx.x & 7, mt = blockIdx.y;
  const float* Pp = W + (((size_t)(nl * 8 + h)) * S + mt * 16) * S;
  const float* Vp = V + ((size_t)nl * S) * 128 + (h >> 1) * 32;
  v8f acc0 = {0,0,0,0,0,0,0,0}, acc1 = {0,0,0,0,0,0,0,0};
  for (int k = 0; k < S; k += 32) {
    v16h a  = load_a_f32(Pp, S, k, lane);
    v16h b0 = load_b_col_f32(Vp,      128, k, lane);
    v16h b1 = load_b_col_f32(Vp + 16, 128, k, lane);
    acc0 = WMMA16(a, b0, acc0);
    acc1 = WMMA16(a, b1, acc1);
  }
  const int rb = (lane >> 4) << 3, col = lane & 15;
  float* Op = O + ((size_t)nl * S + mt * 16) * 256 + h * 32;
#pragma unroll
  for (int r = 0; r < 8; ++r) {
    Op[(size_t)(rb + r) * 256 + col]      = acc0[r];
    Op[(size_t)(rb + r) * 256 + 16 + col] = acc1[r];
  }
}

// Y = LayerNorm(X + A) * g + be. One wave per 256-wide row.
__global__ void __launch_bounds__(256)
add_ln_kernel(const float* __restrict__ X, const float* __restrict__ A,
              const float* __restrict__ g, const float* __restrict__ be,
              float* __restrict__ Y) {
  const int w = threadIdx.x >> 5, lane = threadIdx.x & 31;
  const size_t row = (size_t)blockIdx.x * 8 + w;
  const float* xp = X + row * 256;
  const float* ap = A + row * 256;
  float vals[8], s = 0.f, sq = 0.f;
#pragma unroll
  for (int i = 0; i < 8; ++i) {
    float v = xp[lane + 32 * i] + ap[lane + 32 * i];
    vals[i] = v; s += v; sq += v * v;
  }
  for (int off = 16; off > 0; off >>= 1) { s += __shfl_xor(s, off); sq += __shfl_xor(sq, off); }
  float mean = s * (1.f / 256.f);
  float var  = sq * (1.f / 256.f) - mean * mean;
  float rstd = rsqrtf(var + 1e-5f);
  float* yp = Y + row * 256;
#pragma unroll
  for (int i = 0; i < 8; ++i) {
    int c = lane + 32 * i;
    yp[c] = (vals[i] - mean) * rstd * g[c] + be[c];
  }
}

// Fused FFN + residual + LayerNorm.
// Block = 8 waves, 16 rows. Hidden (16x2048) lives only as 16x32 LDS chunks:
//   stage A: waves cooperatively build H = lrelu(X@W1+b1) chunk (K split over waves)
//   stage B: each wave consumes H as WMMA-A against its 32 output columns of W2.
__global__ void __launch_bounds__(256)
ffn_ln_kernel(const float* __restrict__ X, const _Float16* __restrict__ W1t,
              const float* __restrict__ b1, const _Float16* __restrict__ W2t,
              const float* __restrict__ b2, const float* __restrict__ g,
              const float* __restrict__ be, float* __restrict__ Y) {
  __shared__ _Float16 sX[16 * 256];     // 8 KB  input tile (f16)
  __shared__ float    sPart[8][512];    // 16 KB per-wave WMMA partials
  __shared__ _Float16 sH[512];          // 1 KB  hidden chunk (f16)
  __shared__ float    sOut[16 * 256];   // 16 KB FFN output tile (f32)

  const int tid = threadIdx.x, w = tid >> 5, lane = tid & 31;
  const int rb = (lane >> 4) << 3, col = lane & 15;
  const float* Xp = X + (size_t)blockIdx.x * 16 * 256;

  for (int i = tid; i < 16 * 256; i += 256) sX[i] = (_Float16)Xp[i];
  __syncthreads();

  v8f acc0 = {0,0,0,0,0,0,0,0}, acc1 = {0,0,0,0,0,0,0,0};
  for (int kf = 0; kf < 2048; kf += 32) {
    // ---- stage A: partial for K slice [w*32, w*32+32)
    v16h a  = load_a_h(sX, 256, w * 32, lane);
    v16h b0 = load_b_h(W1t + (size_t)kf * 256,        256, w * 32, lane);
    v16h b1m= load_b_h(W1t + (size_t)(kf + 16) * 256, 256, w * 32, lane);
    v8f p0 = {0,0,0,0,0,0,0,0}, p1 = {0,0,0,0,0,0,0,0};
    p0 = WMMA16(a, b0, p0);
    p1 = WMMA16(a, b1m, p1);
#pragma unroll
    for (int r = 0; r < 8; ++r) {
      sPart[w][(rb + r) * 32 + col]      = p0[r];
      sPart[w][(rb + r) * 32 + 16 + col] = p1[r];
    }
    __syncthreads();
    // reduce 8 partials, bias, LeakyReLU(0.01), convert to f16
    for (int i = tid; i < 512; i += 256) {
      float sum = b1[kf + (i & 31)];
#pragma unroll
      for (int ww = 0; ww < 8; ++ww) sum += sPart[ww][i];
      sum = sum > 0.f ? sum : 0.01f * sum;
      sH[i] = (_Float16)sum;
    }
    __syncthreads();
    // ---- stage B: acc += H @ W2[kf:kf+32, w*32 : w*32+32)
    v16h a2 = load_a_h(sH, 32, 0, lane);
    v16h c0 = load_b_h(W2t + (size_t)(w * 32) * 2048,      2048, kf, lane);
    v16h c1 = load_b_h(W2t + (size_t)(w * 32 + 16) * 2048, 2048, kf, lane);
    acc0 = WMMA16(a2, c0, acc0);
    acc1 = WMMA16(a2, c1, acc1);
  }

  // FFN output (+b2) to LDS, then residual + LayerNorm per row
#pragma unroll
  for (int r = 0; r < 8; ++r) {
    sOut[(rb + r) * 256 + w * 32 + col]      = acc0[r] + b2[w * 32 + col];
    sOut[(rb + r) * 256 + w * 32 + 16 + col] = acc1[r] + b2[w * 32 + 16 + col];
  }
  __syncthreads();
#pragma unroll
  for (int rr = 0; rr < 2; ++rr) {
    int m = w * 2 + rr;
    const float* xr = Xp + (size_t)m * 256;
    float vals[8], s = 0.f, sq = 0.f;
#pragma unroll
    for (int i = 0; i < 8; ++i) {
      int c = lane + 32 * i;
      float v = sOut[m * 256 + c] + xr[c];
      vals[i] = v; s += v; sq += v * v;
    }
    for (int off = 16; off > 0; off >>= 1) { s += __shfl_xor(s, off); sq += __shfl_xor(sq, off); }
    float mean = s * (1.f / 256.f);
    float var  = sq * (1.f / 256.f) - mean * mean;
    float rstd = rsqrtf(var + 1e-5f);
    float* yr = Y + (size_t)blockIdx.x * 16 * 256 + (size_t)m * 256;
#pragma unroll
    for (int i = 0; i < 8; ++i) {
      int c = lane + 32 * i;
      yr[c] = (vals[i] - mean) * rstd * g[c] + be[c];
    }
  }
}

// ---------------------------------------------------------------------------
struct BP {
  const _Float16 *wq, *wk, *wv, *wo, *w1, *w2;
  const float *bq, *bk, *bv, *bo, *g1, *be1, *b1, *b2, *g2, *be2;
};

static void run_block(hipStream_t s, const float* Xin, float* q, float* k, float* v,
                      float* o, float* a, float* x1, float* Xout, float* Wout,
                      int Sloc, int S, const BP& P) {
  const int M = Sloc * S;  // 8192 tokens per chunk
  dim3 wv1(32);
  gemm_bias_kernel<<<dim3(M / 16, 4), wv1, 0, s>>>(Xin, P.wq, P.bq, q, 256, 256);
  gemm_bias_kernel<<<dim3(M / 16, 2), wv1, 0, s>>>(Xin, P.wk, P.bk, k, 256, 128);
  gemm_bias_kernel<<<dim3(M / 16, 2), wv1, 0, s>>>(Xin, P.wv, P.bv, v, 256, 128);
  attn_scores_kernel<<<dim3(Sloc * 8, S / 16), wv1, 0, s>>>(q, k, Wout, S);
  softmax_kernel<<<dim3(Sloc * 8 * S / 8), dim3(256), 0, s>>>(Wout, S);
  attn_out_kernel<<<dim3(Sloc * 8, S / 16), wv1, 0, s>>>(Wout, v, o, S);
  gemm_bias_kernel<<<dim3(M / 16, 4), wv1, 0, s>>>(o, P.wo, P.bo, a, 256, 256);
  add_ln_kernel<<<dim3(M / 8), dim3(256), 0, s>>>(Xin, a, P.g1, P.be1, x1);
  ffn_ln_kernel<<<dim3(M / 16), dim3(256), 0, s>>>(x1, P.w1, P.b1, P.w2, P.b2, P.g2, P.be2, Xout);
}

extern "C" void kernel_launch(void* const* d_in, const int* in_sizes, int n_in,
                              void* d_out, int out_size, void* d_ws, size_t ws_size,
                              hipStream_t stream) {
  (void)in_sizes; (void)n_in; (void)out_size; (void)ws_size;
  const float* x = (const float*)d_in[0];

  // Workspace layout
  char* base = (char*)d_ws;
  size_t off = 0;
  auto allocH = [&](size_t n) { _Float16* p = (_Float16*)(base + off); off += n * sizeof(_Float16); return p; };
  auto allocF = [&](size_t n) { float* p = (float*)(base + off); off += n * sizeof(float); return p; };

  _Float16* fwq = allocH(256 * 256);  _Float16* fwk = allocH(256 * 128);
  _Float16* fwv = allocH(256 * 128);  _Float16* fwo = allocH(256 * 256);
  _Float16* fw1 = allocH(256 * 2048); _Float16* fw2 = allocH(2048 * 256);
  _Float16* twq = allocH(256 * 256);  _Float16* twk = allocH(256 * 128);
  _Float16* twv = allocH(256 * 128);  _Float16* two = allocH(256 * 256);
  _Float16* tw1 = allocH(256 * 2048); _Float16* tw2 = allocH(2048 * 256);

  const size_t CH = 8192 * 256;  // per-chunk token block
  float* xf = allocF(CH);  float* xt = allocF(CH);
  float* q  = allocF(CH);  float* kb = allocF(CH / 2);
  float* vb = allocF(CH / 2);
  float* ob = allocF(CH);  float* ab = allocF(CH);
  float* x1 = allocF(CH);  float* x2 = allocF(CH);

  // Convert + transpose weights to f16 (N,K)
  auto tw = [&](const float* W, _Float16* Wt, int K, int N) {
    transpose_w_kernel<<<dim3((K * N + 255) / 256), dim3(256), 0, stream>>>(W, Wt, K, N);
  };
  tw((const float*)d_in[1], fwq, 256, 256);
  tw((const float*)d_in[3], fwk, 256, 128);
  tw((const float*)d_in[5], fwv, 256, 128);
  tw((const float*)d_in[7], fwo, 256, 256);
  tw((const float*)d_in[25], fw1, 256, 2048);
  tw((const float*)d_in[27], fw2, 2048, 256);
  tw((const float*)d_in[9],  twq, 256, 256);
  tw((const float*)d_in[11], twk, 256, 128);
  tw((const float*)d_in[13], twv, 256, 128);
  tw((const float*)d_in[15], two, 256, 256);
  tw((const float*)d_in[29], tw1, 256, 2048);
  tw((const float*)d_in[31], tw2, 2048, 256);

  BP featP = { fwq, fwk, fwv, fwo, fw1, fw2,
               (const float*)d_in[2],  (const float*)d_in[4],
               (const float*)d_in[6],  (const float*)d_in[8],
               (const float*)d_in[17], (const float*)d_in[18],
               (const float*)d_in[26], (const float*)d_in[28],
               (const float*)d_in[19], (const float*)d_in[20] };
  BP timeP = { twq, twk, twv, two, tw1, tw2,
               (const float*)d_in[10], (const float*)d_in[12],
               (const float*)d_in[14], (const float*)d_in[16],
               (const float*)d_in[21], (const float*)d_in[22],
               (const float*)d_in[30], (const float*)d_in[32],
               (const float*)d_in[23], (const float*)d_in[24] };

  float* out0   = (float*)d_out;                 // (B,F,T,D)
  float* feat_w = out0 + (size_t)16777216;       // (B*T, 8, 64, 64)
  float* time_w = feat_w + (size_t)33554432;     // (B*F, 8, 128, 128)

  for (int b = 0; b < 8; ++b) {
    // ---- feature attention block: 128 sequences of length 64
    gather_feat_kernel<<<dim3(8192), dim3(256), 0, stream>>>(x, xf, b);
    run_block(stream, xf, q, kb, vb, ob, ab, x1, x2,
              feat_w + (size_t)b * 128 * 8 * 64 * 64, /*Sloc=*/128, /*S=*/64, featP);
    scatter_feat_kernel<<<dim3(8192), dim3(256), 0, stream>>>(x2, xt);
    // ---- time attention block: 64 sequences of length 128, writes final output
    run_block(stream, xt, q, kb, vb, ob, ab, x1,
              out0 + (size_t)b * 8192 * 256,
              time_w + (size_t)b * 64 * 8 * 128 * 128, /*Sloc=*/64, /*S=*/128, timeP);
  }
}
